// TargetRegion_74569222193153
// MI455X (gfx1250) — matrix operand order repaired and checked by reading, not compiled, not run on hardware
//
#include <hip/hip_runtime.h>
#include <math.h>

// ---------------------------------------------------------------------------
// Problem constants (from the reference)
// ---------------------------------------------------------------------------
#define MM   6
#define NN   96
#define PP   384
#define DD   64
#define HH   8
#define MN   576            // MM*NN
#define TT   30
#define TSS  15
#define TILES 24            // PP/16
#define WAVES 4
#define TPB  128

typedef __attribute__((ext_vector_type(16))) __bf16 v16bf;
typedef __attribute__((ext_vector_type(8)))  float  v8f;

#define WMMA_BF16(A,B,C) \
  __builtin_amdgcn_wmma_f32_16x16x32_bf16(false,(A),false,(B),(short)0,(C),false,false)

// Gather a 16x32 bf16 A-fragment (per ISA 7.12.2 layout) from a row-major
// f32 LDS buffer row.  lane -> M = lane&15, half = lane>>4 selects K-subrange.
__device__ __forceinline__ v16bf gatherA(const float* rp, int half, int koff) {
  v16bf a;
#pragma unroll
  for (int j = 0; j < 16; ++j) {
    int kl = (j < 8) ? (half * 8 + j) : (16 + half * 8 + (j - 8));
    a[j] = (__bf16)rp[koff + kl];
  }
  return a;
}

// ---------------------------------------------------------------------------
// Encoder parameter bundle (one AL-encoder block)
// ---------------------------------------------------------------------------
struct EncP {
  const float *lane_l1w, *lane_l1b, *lane_lng, *lane_lnb, *lane_l2w, *lane_l2b;
  const float *edge_l1w, *edge_l1b, *edge_lng, *edge_lnb, *edge_l2w, *edge_l2b;
  const float *ao_l1w,   *ao_l1b,   *ao_lng,   *ao_lnb,   *ao_l2w,   *ao_l2b;
  const float *tab_int, *tab_turn, *tab_tc;
  const float *ln1g, *ln1b, *aggr_w, *aggr_b, *ln2g, *ln2b;
  const float *qw, *qb, *kw, *kb, *vw, *vb;
  const float *selfw, *selfb, *ihw, *ihb, *hhw, *hhb;
  const float *outw, *outb;
  const float *n1g, *n1b, *n2g, *n2b;
  const float *m1w, *m1b, *m2w, *m2b;
};

// ---------------------------------------------------------------------------
// Fused AL-encoder: one block per (m,n); 4 waves, WMMA bf16 GEMMs in LDS.
// Weight matrices arrive via GLOBAL_LOAD_ASYNC_TO_LDS (ASYNCcnt-tracked DMA),
// then get repacked in-LDS into WMMA B-fragment layout.
// ---------------------------------------------------------------------------
__global__ __launch_bounds__(TPB)
void enc_kernel(EncP P, float* ego, const float* geom, int s,
                const float* lane_pos, const float* lane_vec,
                const int* iint, const int* iturn, const int* itc) {
  const int b    = blockIdx.x;
  const int n    = b % NN;
  const int tid  = threadIdx.x;
  const int wave = tid >> 5;
  const int lane = tid & 31;
  const int row  = lane & 15;
  const int half = lane >> 4;

  // B-fragments for the 5 64x64 GEMMs: [gemm][kTile][nTile][lane][16 bf16]
  __shared__ __attribute__((aligned(32))) __bf16 s_wfrag[5][2][4][32][16];
  __shared__ __attribute__((aligned(16))) float s_wraw[64 * 64]; // async DMA scratch
  __shared__ float s_buf[WAVES][2][16][64];         // per-wave staging tiles
  __shared__ float s_l1w[2][128], s_l1b[2][64], s_elng[2][64], s_elnb[2][64];
  __shared__ float s_colc[64], s_tab[7][64];
  __shared__ float s_ln1g[64], s_ln1b[64], s_ln2g[64], s_ln2b[64];
  __shared__ float s_aggb[64], s_kb[64], s_vb[64];
  __shared__ float s_xa[64], s_q[64], s_ego[64];
  __shared__ unsigned s_cls[WAVES][16];
  __shared__ float s_part[WAVES][HH][10];
  __shared__ float s_msg[64], s_tmp[64], s_tmp2[64], s_mlp[256];

  const float* g8 = geom + (b * 2 + s) * 8;
  const float tgx = g8[0], tgy = g8[1], rc = g8[2], rsn = g8[3];
  const float aox = g8[4], aoy = g8[5], dis = g8[6];

  // ---- stage small parameters ----
  for (int i = tid; i < 128; i += TPB) {
    s_l1w[0][i] = P.lane_l1w[i];
    s_l1w[1][i] = P.edge_l1w[i];
  }
  for (int i = tid; i < 64; i += TPB) {
    s_l1b[0][i] = P.lane_l1b[i];  s_l1b[1][i] = P.edge_l1b[i];
    s_elng[0][i] = P.lane_lng[i]; s_elnb[0][i] = P.lane_lnb[i];
    s_elng[1][i] = P.edge_lng[i]; s_elnb[1][i] = P.edge_lnb[i];
    s_ln1g[i] = P.ln1g[i]; s_ln1b[i] = P.ln1b[i];
    s_ln2g[i] = P.ln2g[i]; s_ln2b[i] = P.ln2b[i];
    s_aggb[i] = P.aggr_b[i]; s_kb[i] = P.kb[i]; s_vb[i] = P.vb[i];
    s_ego[i] = ego[b * 64 + i];
  }
  for (int i = tid; i < 7 * 64; i += TPB) {
    float v = (i < 128) ? P.tab_int[i] : (i < 320) ? P.tab_turn[i - 128] : P.tab_tc[i - 320];
    s_tab[i >> 6][i & 63] = v;
  }

  // ---- async-DMA each 64x64 weight matrix to LDS, then repack to fragments ----
  {
    const float* gs[5] = { P.lane_l2w, P.edge_l2w, P.aggr_w, P.kw, P.vw };
    unsigned lds_base = (unsigned)(unsigned long long)(void*)&s_wraw[0];
    for (int g = 0; g < 5; ++g) {
      const float* src = gs[g];
#pragma unroll
      for (int it = 0; it < 8; ++it) {
        unsigned byteOff = (unsigned)((tid + it * TPB) * 16);   // 16B per b128 xfer
        unsigned dst = lds_base + byteOff;
        asm volatile("global_load_async_to_lds_b128 %0, %1, %2"
                     :: "v"(dst), "v"(byteOff), "s"(src)
                     : "memory");
      }
      asm volatile("s_wait_asynccnt 0x0" ::: "memory");
      __syncthreads();
      for (int idx = tid; idx < 2 * 4 * 32 * 16; idx += TPB) {
        int j  = idx & 15;
        int ln = (idx >> 4) & 31;
        int nc = (idx >> 9) & 3;
        int kc = (idx >> 11) & 1;
        int col = nc * 16 + (ln & 15);
        int k   = kc * 32 + ((j < 8) ? ((ln >> 4) * 8 + j) : (16 + (ln >> 4) * 8 + (j - 8)));   // B 32x16 layout, table k order (matches gatherA)
        s_wfrag[g][kc][nc][ln][j] = (__bf16)s_wraw[k * 64 + col];
      }
      __syncthreads();
    }
  }

  // ---- xa = ln(norm1, ego); q = lin_q(xa) ----
  if (tid < 64) {
    float mu = 0.f, m2 = 0.f;
    for (int k = 0; k < 64; ++k) mu += s_ego[k];
    mu *= (1.f / 64.f);
    for (int k = 0; k < 64; ++k) { float d = s_ego[k] - mu; m2 += d * d; }
    m2 *= (1.f / 64.f);
    s_xa[tid] = (s_ego[tid] - mu) * rsqrtf(m2 + 1e-5f) * P.n1g[tid] + P.n1b[tid];
  }
  __syncthreads();
  if (tid < 64) {
    float a = P.qb[tid];
    for (int k = 0; k < 64; ++k) a += s_xa[k] * P.qw[k * 64 + tid];
    s_q[tid] = a;
    // ao embedding stage 1: rotate vec_ao then l1 (2->64)
    float r0 =  aox * rc + aoy * rsn;
    float r1 = -aox * rsn + aoy * rc;
    s_tmp[tid] = r0 * P.ao_l1w[tid] + r1 * P.ao_l1w[64 + tid] + P.ao_l1b[tid];
  }
  __syncthreads();
  if (tid < 64) {
    float mu = 0.f, m2 = 0.f;
    for (int k = 0; k < 64; ++k) { mu += s_tmp[k]; m2 += s_tmp[k] * s_tmp[k]; }
    mu *= (1.f / 64.f);
    float var = m2 * (1.f / 64.f) - mu * mu;
    s_tmp2[tid] = fmaxf(0.f, (s_tmp[tid] - mu) * rsqrtf(fmaxf(var, 0.f) + 1e-5f) * P.ao_lng[tid] + P.ao_lnb[tid]);
  }
  __syncthreads();
  if (tid < 64) {
    float a = P.ao_l2b[tid] + P.lane_l2b[tid] + P.edge_l2b[tid];
    for (int k = 0; k < 64; ++k) a += s_tmp2[k] * P.ao_l2w[k * 64 + tid];
    s_colc[tid] = a;
  }
  __syncthreads();

  // ---- per-wave streaming over P in 16-row tiles ----
  float run_mx = -3.0e38f, run_sum = 0.f;
  float run_macc[8] = {0.f,0.f,0.f,0.f,0.f,0.f,0.f,0.f};

  for (int tile = wave; tile < TILES; tile += WAVES) {
    const int p = tile * 16 + row;

    // prefetch next tile's point stream into cache (global_prefetch_b8)
    {
      int pn = (tile + WAVES) * 16 + row;
      if (pn < PP) {
        __builtin_prefetch(&lane_pos[(n * PP + pn) * 2], 0, 1);
        __builtin_prefetch(&lane_vec[(n * PP + pn) * 2], 0, 1);
      }
    }

    // Phase 1: per-lane point embedding (half 0 = lane path, half 1 = edge path)
    float x0, x1;
    if (half == 0) {
      x0 = lane_vec[(n * PP + p) * 2];
      x1 = lane_vec[(n * PP + p) * 2 + 1];
    } else {
      x0 = lane_pos[(n * PP + p) * 2]     - tgx;
      x1 = lane_pos[(n * PP + p) * 2 + 1] - tgy;
      unsigned mk = (sqrtf(x0 * x0 + x1 * x1) < dis) ? 1u : 0u;
      s_cls[wave][row] = (unsigned)iint[n * PP + p] | ((unsigned)iturn[n * PP + p] << 1)
                       | ((unsigned)itc[n * PP + p] << 3) | (mk << 4);
    }
    float r0 =  x0 * rc + x1 * rsn;
    float r1 = -x0 * rsn + x1 * rc;
    {
      float mu = 0.f, m2 = 0.f;
      for (int j = 0; j < 64; ++j) {
        float t = r0 * s_l1w[half][j] + r1 * s_l1w[half][64 + j] + s_l1b[half][j];
        mu += t; m2 += t * t;
      }
      mu *= (1.f / 64.f);
      float var = m2 * (1.f / 64.f) - mu * mu;
      float rstd = rsqrtf(fmaxf(var, 0.f) + 1e-5f);
      for (int j = 0; j < 64; ++j) {
        float t = r0 * s_l1w[half][j] + r1 * s_l1w[half][64 + j] + s_l1b[half][j];
        s_buf[wave][half][row][j] = fmaxf(0.f, (t - mu) * rstd * s_elng[half][j] + s_elnb[half][j]);
      }
    }

    // Phase 2: e = emb_lane.l2 @ uL + emb_edge.l2 @ uE   (16 WMMAs)
    v16bf aL0 = gatherA(&s_buf[wave][0][row][0], half, 0);
    v16bf aL1 = gatherA(&s_buf[wave][0][row][0], half, 32);
    v16bf aE0 = gatherA(&s_buf[wave][1][row][0], half, 0);
    v16bf aE1 = gatherA(&s_buf[wave][1][row][0], half, 32);
    v8f acc[4] = {};
#pragma unroll
    for (int nc = 0; nc < 4; ++nc) {
      v16bf b0 = *(const v16bf*)(&s_wfrag[0][0][nc][lane][0]);
      v16bf b1 = *(const v16bf*)(&s_wfrag[0][1][nc][lane][0]);
      v16bf c0 = *(const v16bf*)(&s_wfrag[1][0][nc][lane][0]);
      v16bf c1 = *(const v16bf*)(&s_wfrag[1][1][nc][lane][0]);
      acc[nc] = WMMA_BF16(aL0, b0, acc[nc]);
      acc[nc] = WMMA_BF16(aL1, b1, acc[nc]);
      acc[nc] = WMMA_BF16(aE0, c0, acc[nc]);
      acc[nc] = WMMA_BF16(aE1, c1, acc[nc]);
    }
    // fixup: biases + ao emb + category tables, dump to buf0
#pragma unroll
    for (int nc = 0; nc < 4; ++nc) {
      int col = nc * 16 + row;
#pragma unroll
      for (int i = 0; i < 8; ++i) {
        int rr = i + 8 * half;
        unsigned cw = s_cls[wave][rr];
        float add = s_colc[col] + s_tab[cw & 1][col] + s_tab[2 + ((cw >> 1) & 3)][col]
                  + s_tab[5 + ((cw >> 3) & 1)][col];
        s_buf[wave][0][rr][col] = acc[nc][i] + add;
      }
    }

    // Phase 3: ln1 + relu (rows owned by lanes 0-15)
    if (half == 0) {
      float mu = 0.f, m2 = 0.f;
      for (int j = 0; j < 64; ++j) { float t = s_buf[wave][0][row][j]; mu += t; m2 += t * t; }
      mu *= (1.f / 64.f);
      float var = m2 * (1.f / 64.f) - mu * mu;
      float rstd = rsqrtf(fmaxf(var, 0.f) + 1e-5f);
      for (int j = 0; j < 64; ++j) {
        float t = s_buf[wave][0][row][j];
        s_buf[wave][1][row][j] = fmaxf(0.f, (t - mu) * rstd * s_ln1g[j] + s_ln1b[j]);
      }
    }

    // Phase 4: aggr_lin WMMA, then ln2
    {
      v16bf f0 = gatherA(&s_buf[wave][1][row][0], half, 0);
      v16bf f1 = gatherA(&s_buf[wave][1][row][0], half, 32);
      v8f ac2[4] = {};
#pragma unroll
      for (int nc = 0; nc < 4; ++nc) {
        v16bf b0 = *(const v16bf*)(&s_wfrag[2][0][nc][lane][0]);
        v16bf b1 = *(const v16bf*)(&s_wfrag[2][1][nc][lane][0]);
        ac2[nc] = WMMA_BF16(f0, b0, ac2[nc]);
        ac2[nc] = WMMA_BF16(f1, b1, ac2[nc]);
      }
#pragma unroll
      for (int nc = 0; nc < 4; ++nc) {
        int col = nc * 16 + row;
#pragma unroll
        for (int i = 0; i < 8; ++i)
          s_buf[wave][0][i + 8 * half][col] = ac2[nc][i] + s_aggb[col];
      }
    }
    if (half == 0) {
      float mu = 0.f, m2 = 0.f;
      for (int j = 0; j < 64; ++j) { float t = s_buf[wave][0][row][j]; mu += t; m2 += t * t; }
      mu *= (1.f / 64.f);
      float var = m2 * (1.f / 64.f) - mu * mu;
      float rstd = rsqrtf(fmaxf(var, 0.f) + 1e-5f);
      for (int j = 0; j < 64; ++j) {
        float t = s_buf[wave][0][row][j];
        s_buf[wave][1][row][j] = (t - mu) * rstd * s_ln2g[j] + s_ln2b[j];
      }
    }

    // Phase 5: k = lin_k(e), v = lin_v(e)  (8 + 8 WMMAs)
    {
      v16bf e0 = gatherA(&s_buf[wave][1][row][0], half, 0);
      v16bf e1 = gatherA(&s_buf[wave][1][row][0], half, 32);
      v8f ack[4] = {}, acv[4] = {};
#pragma unroll
      for (int nc = 0; nc < 4; ++nc) {
        v16bf kb0 = *(const v16bf*)(&s_wfrag[3][0][nc][lane][0]);
        v16bf kb1 = *(const v16bf*)(&s_wfrag[3][1][nc][lane][0]);
        v16bf vb0 = *(const v16bf*)(&s_wfrag[4][0][nc][lane][0]);
        v16bf vb1 = *(const v16bf*)(&s_wfrag[4][1][nc][lane][0]);
        ack[nc] = WMMA_BF16(e0, kb0, ack[nc]);
        ack[nc] = WMMA_BF16(e1, kb1, ack[nc]);
        acv[nc] = WMMA_BF16(e0, vb0, acv[nc]);
        acv[nc] = WMMA_BF16(e1, vb1, acv[nc]);
      }
#pragma unroll
      for (int nc = 0; nc < 4; ++nc) {
        int col = nc * 16 + row;
#pragma unroll
        for (int i = 0; i < 8; ++i) {
          s_buf[wave][0][i + 8 * half][col] = ack[nc][i] + s_kb[col];
          s_buf[wave][1][i + 8 * half][col] = acv[nc][i] + s_vb[col];
        }
      }
    }

    // Phase 6: masked online softmax per head (lanes 0-7 own heads)
    if (lane < 8) {
      const int h = lane;
      for (int rr = 0; rr < 16; ++rr) {
        unsigned cw = s_cls[wave][rr];
        if ((cw >> 4) & 1u) {
          float lg = 0.f;
#pragma unroll
          for (int d = 0; d < 8; ++d) lg += s_q[h * 8 + d] * s_buf[wave][0][rr][h * 8 + d];
          lg *= 0.35355339059327373f;          // 1/sqrt(DH)
          float nm = fmaxf(run_mx, lg);
          float sc = expf(run_mx - nm);
          float w  = expf(lg - nm);
          run_sum = run_sum * sc + w;
#pragma unroll
          for (int d = 0; d < 8; ++d)
            run_macc[d] = run_macc[d] * sc + w * s_buf[wave][1][rr][h * 8 + d];
          run_mx = nm;
        }
      }
    }
  } // tile loop

  if (lane < 8) {
    s_part[wave][lane][0] = run_mx;
    s_part[wave][lane][1] = run_sum;
#pragma unroll
    for (int d = 0; d < 8; ++d) s_part[wave][lane][2 + d] = run_macc[d];
  }
  __syncthreads();

  // ---- merge per-wave softmax partials -> msg ----
  if (tid < 8) {
    float mx = -3.0e38f, sum = 0.f, mac[8] = {0.f,0.f,0.f,0.f,0.f,0.f,0.f,0.f};
    for (int w2 = 0; w2 < WAVES; ++w2) {
      float pmx = s_part[w2][tid][0], psum = s_part[w2][tid][1];
      float nm = fmaxf(mx, pmx);
      float s1 = expf(mx - nm), s2 = expf(pmx - nm);
      sum = sum * s1 + psum * s2;
#pragma unroll
      for (int d = 0; d < 8; ++d) mac[d] = mac[d] * s1 + s_part[w2][tid][2 + d] * s2;
      mx = nm;
    }
    float inv = 1.f / fmaxf(sum, 1e-9f);
#pragma unroll
    for (int d = 0; d < 8; ++d) s_msg[tid * 8 + d] = mac[d] * inv;
  }
  __syncthreads();

  // ---- gating + out + MLP (per-row matvecs) ----
  if (tid < 64) {
    float msgv = s_msg[tid];
    float a1 = P.ihb[tid], a2 = P.hhb[tid], a3 = P.selfb[tid];
    for (int k = 0; k < 64; ++k) {
      a1 += s_msg[k] * P.ihw[k * 64 + tid];
      a2 += s_xa[k]  * P.hhw[k * 64 + tid];
      a3 += s_xa[k]  * P.selfw[k * 64 + tid];
    }
    float g = 1.f / (1.f + expf(-(a1 + a2)));
    s_tmp[tid] = msgv + g * (a3 - msgv);               // upd
  }
  __syncthreads();
  if (tid < 64) {
    float o = P.outb[tid];
    for (int k = 0; k < 64; ++k) o += s_tmp[k] * P.outw[k * 64 + tid];
    s_tmp2[tid] = s_ego[tid] + o;                      // x1 residual
  }
  __syncthreads();
  if (tid < 64) {
    float mu = 0.f, m2 = 0.f;
    for (int k = 0; k < 64; ++k) { float t = s_tmp2[k]; mu += t; m2 += t * t; }
    mu *= (1.f / 64.f);
    float var = m2 * (1.f / 64.f) - mu * mu;
    s_tmp[tid] = (s_tmp2[tid] - mu) * rsqrtf(fmaxf(var, 0.f) + 1e-5f) * P.n2g[tid] + P.n2b[tid];
  }
  __syncthreads();
  for (int jj = tid; jj < 256; jj += TPB) {
    float a = P.m1b[jj];
    for (int k = 0; k < 64; ++k) a += s_tmp[k] * P.m1w[k * 256 + jj];
    s_mlp[jj] = fmaxf(0.f, a);
  }
  __syncthreads();
  if (tid < 64) {
    float a = P.m2b[tid];
    for (int k = 0; k < 256; ++k) a += s_mlp[k] * P.m2w[k * 64 + tid];
    ego[b * 64 + tid] = s_tmp2[tid] + a;
  }
}

// ---------------------------------------------------------------------------
// ego0 = ego_embed @ fc.w + fc.b   (576 x 128 -> 576 x 64)
// ---------------------------------------------------------------------------
__global__ __launch_bounds__(64)
void fc_kernel(const float* x, const float* w, const float* bias,
               float* ego, float* emb0) {
  __shared__ float s_x[128];
  int b = blockIdx.x, j = threadIdx.x;
  s_x[j] = x[b * 128 + j];
  s_x[64 + j] = x[b * 128 + 64 + j];
  __syncthreads();
  float a = bias[j];
  for (int k = 0; k < 128; ++k) a += s_x[k] * w[k * 64 + j];
  ego[b * 64 + j]  = a;
  emb0[b * 64 + j] = a;
}

// ---------------------------------------------------------------------------
// Per-(m,n) geometry: y_glob targets, heading rot, vec_ao, clipped radius.
// ---------------------------------------------------------------------------
__global__ void geom_kernel(const float* yinit, const float* rotl, const float* orig,
                            float radii, float* geom) {
  int b = blockIdx.x * blockDim.x + threadIdx.x;
  if (b >= MN) return;
  int n = b % NN;
  const float* Y = yinit + b * (TT * 2);
  float RL00 = rotl[n*4+0], RL01 = rotl[n*4+1], RL10 = rotl[n*4+2], RL11 = rotl[n*4+3];
  float ox = orig[n*2], oy = orig[n*2+1];
  float dis = 0.f, px = 0.f, py = 0.f;
  float g13x=0,g13y=0,g14x=0,g14y=0,g28x=0,g28y=0,g29x=0,g29y=0;
  for (int t = 0; t < TT; ++t) {
    float a0 = Y[t*2], a1 = Y[t*2+1];
    float dx = a0 - px, dy = a1 - py;
    dis += sqrtf(dx*dx + dy*dy);
    px = a0; py = a1;
    float gx = a0*RL00 + a1*RL10 + ox;
    float gy = a0*RL01 + a1*RL11 + oy;
    if (t == 13) { g13x=gx; g13y=gy; } else if (t == 14) { g14x=gx; g14y=gy; }
    else if (t == 28) { g28x=gx; g28y=gy; } else if (t == 29) { g29x=gx; g29y=gy; }
  }
  dis *= radii;
  dis = fminf(fmaxf(dis, 2.0f), 10.0f);
  {
    float th = atan2f(g14y - g13y, g14x - g13x);
    float c = cosf(th), sn = sinf(th);
    float* o8 = geom + (b*2 + 0) * 8;
    o8[0]=g14x; o8[1]=g14y; o8[2]=c; o8[3]=sn; o8[4]=ox-g14x; o8[5]=oy-g14y; o8[6]=dis; o8[7]=0.f;
  }
  {
    float th = atan2f(g29y - g28y, g29x - g28x);
    float c = cosf(th), sn = sinf(th);
    float* o8 = geom + (b*2 + 1) * 8;
    o8[0]=g29x; o8[1]=g29y; o8[2]=c; o8[3]=sn; o8[4]=ox-g29x; o8[5]=oy-g29y; o8[6]=dis; o8[7]=0.f;
  }
}

// ---------------------------------------------------------------------------
// GRU over embed sequence + score head (1 scalar per row)
// ---------------------------------------------------------------------------
__global__ __launch_bounds__(64)
void gru_score_kernel(const float* emb, int Ls,
                      const float* w_ih, const float* w_hh,
                      const float* b_ih, const float* b_hh,
                      const float* l1w, const float* l1b,
                      const float* lng, const float* lnb,
                      const float* l2w, const float* l2b,
                      const float* pe, float* out) {
  __shared__ float s_h[64], s_x[64], s_u[64];
  int b = blockIdx.x, j = threadIdx.x;
  s_h[j] = 0.f;
  __syncthreads();
  for (int t = 0; t < Ls; ++t) {
    s_x[j] = emb[t * MN * 64 + b * 64 + j];
    __syncthreads();
    float gir = b_ih[j], giz = b_ih[64 + j], gin = b_ih[128 + j];
    float ghr = b_hh[j], ghz = b_hh[64 + j], ghn = b_hh[128 + j];
    for (int k = 0; k < 64; ++k) {
      float x = s_x[k], h = s_h[k];
      gir += x * w_ih[k*192 + j];      giz += x * w_ih[k*192 + 64 + j];  gin += x * w_ih[k*192 + 128 + j];
      ghr += h * w_hh[k*192 + j];      ghz += h * w_hh[k*192 + 64 + j];  ghn += h * w_hh[k*192 + 128 + j];
    }
    float r  = 1.f / (1.f + expf(-(gir + ghr)));
    float z  = 1.f / (1.f + expf(-(giz + ghz)));
    float nn = tanhf(gin + r * ghn);
    float hn = (1.f - z) * nn + z * s_h[j];
    __syncthreads();
    s_h[j] = hn;
    __syncthreads();
  }
  // score head
  s_x[j] = s_h[j] + pe[j];
  __syncthreads();
  float a = l1b[j];
  for (int k = 0; k < 64; ++k) a += s_x[k] * l1w[k * 64 + j];
  s_u[j] = a;
  __syncthreads();
  float mu = 0.f, m2 = 0.f;
  for (int k = 0; k < 64; ++k) { mu += s_u[k]; m2 += s_u[k] * s_u[k]; }
  mu *= (1.f / 64.f);
  float var = m2 * (1.f / 64.f) - mu * mu;
  float un = fmaxf(0.f, (s_u[j] - mu) * rsqrtf(fmaxf(var, 0.f) + 1e-5f) * lng[j] + lnb[j]);
  __syncthreads();
  s_x[j] = un;
  __syncthreads();
  if (j == 0) {
    float o = l2b[0];
    for (int k = 0; k < 64; ++k) o += s_x[k] * l2w[k];
    out[b] = o;
  }
}

// ---------------------------------------------------------------------------
// mlp_head with scalar output (pi head)
// ---------------------------------------------------------------------------
__global__ __launch_bounds__(64)
void head1_kernel(const float* ego, const float* pe,
                  const float* l1w, const float* l1b,
                  const float* lng, const float* lnb,
                  const float* l2w, const float* l2b, float* out) {
  __shared__ float s_x[64], s_u[64];
  int b = blockIdx.x, j = threadIdx.x;
  s_x[j] = ego[b * 64 + j] + pe[j];
  __syncthreads();
  float a = l1b[j];
  for (int k = 0; k < 64; ++k) a += s_x[k] * l1w[k * 64 + j];
  s_u[j] = a;
  __syncthreads();
  float mu = 0.f, m2 = 0.f;
  for (int k = 0; k < 64; ++k) { mu += s_u[k]; m2 += s_u[k] * s_u[k]; }
  mu *= (1.f / 64.f);
  float var = m2 * (1.f / 64.f) - mu * mu;
  float un = fmaxf(0.f, (s_u[j] - mu) * rsqrtf(fmaxf(var, 0.f) + 1e-5f) * lng[j] + lnb[j]);
  __syncthreads();
  s_x[j] = un;
  __syncthreads();
  if (j == 0) {
    float o = l2b[0];
    for (int k = 0; k < 64; ++k) o += s_x[k] * l2w[k];
    out[b] = o;
  }
}

// ---------------------------------------------------------------------------
// dec head: 60 outputs per row -> trajs + delta
// ---------------------------------------------------------------------------
__global__ __launch_bounds__(64)
void dec_kernel(const float* ego, const float* pe,
                const float* l1w, const float* l1b,
                const float* lng, const float* lnb,
                const float* l2w, const float* l2b,
                float* traj_out, float* delta, int s) {
  __shared__ float s_x[64], s_u[64];
  int b = blockIdx.x, j = threadIdx.x;
  s_x[j] = ego[b * 64 + j] + pe[j];
  __syncthreads();
  float a = l1b[j];
  for (int k = 0; k < 64; ++k) a += s_x[k] * l1w[k * 64 + j];
  s_u[j] = a;
  __syncthreads();
  float mu = 0.f, m2 = 0.f;
  for (int k = 0; k < 64; ++k) { mu += s_u[k]; m2 += s_u[k] * s_u[k]; }
  mu *= (1.f / 64.f);
  float var = m2 * (1.f / 64.f) - mu * mu;
  float un = fmaxf(0.f, (s_u[j] - mu) * rsqrtf(fmaxf(var, 0.f) + 1e-5f) * lng[j] + lnb[j]);
  __syncthreads();
  s_x[j] = un;
  __syncthreads();
  if (j < 60) {
    float o = l2b[j];
    for (int k = 0; k < 64; ++k) o += s_x[k] * l2w[k * 60 + j];
    int t = s * TSS + (j >> 2), c = j & 3;
    traj_out[(b * TT + t) * 4 + c] = o;
    if (c < 2) delta[(b * TT + t) * 2 + c] = o;
  }
}

__global__ void copy_kernel(const float* src, float* dst, int nElem) {
  int i = blockIdx.x * blockDim.x + threadIdx.x;
  if (i < nElem) dst[i] = src[i];
}

__global__ void axpy_kernel(float* y, const float* d, int nElem) {
  int i = blockIdx.x * blockDim.x + threadIdx.x;
  if (i < nElem) y[i] += d[i];
}

// ---------------------------------------------------------------------------
// Host: wire up the flattened parameter pytree and sequence the pipeline.
// Input order: 9 data arrays, then params tree-flattened (sorted dict keys):
//   dec(9-14) enc[0](15-63) enc[1](64-112) fc(113-114) gru(115-118)
//   pi(119-124) pos_embed(125) score(126-131)
// ---------------------------------------------------------------------------
static EncP make_enc(void* const* d_in, int base) {
  auto F = [&](int o) { return (const float*)d_in[base + o]; };
  EncP p;
  p.aggr_b  = F(0);  p.aggr_w  = F(1);
  p.ln1b    = F(2);  p.ln1g    = F(3);
  p.ln2b    = F(4);  p.ln2g    = F(5);
  p.ao_l1b  = F(6);  p.ao_l1w  = F(7);  p.ao_l2b  = F(8);  p.ao_l2w  = F(9);
  p.ao_lnb  = F(10); p.ao_lng  = F(11);
  p.edge_l1b= F(12); p.edge_l1w= F(13); p.edge_l2b= F(14); p.edge_l2w= F(15);
  p.edge_lnb= F(16); p.edge_lng= F(17);
  p.lane_l1b= F(18); p.lane_l1w= F(19); p.lane_l2b= F(20); p.lane_l2w= F(21);
  p.lane_lnb= F(22); p.lane_lng= F(23);
  p.tab_int = F(24);
  p.hhb = F(25); p.hhw = F(26);
  p.ihb = F(27); p.ihw = F(28);
  p.kb  = F(29); p.kw  = F(30);
  p.qb  = F(31); p.qw  = F(32);
  p.selfb = F(33); p.selfw = F(34);
  p.vb  = F(35); p.vw  = F(36);
  p.m1b = F(37); p.m1w = F(38); p.m2b = F(39); p.m2w = F(40);
  p.n1b = F(41); p.n1g = F(42); p.n2b = F(43); p.n2g = F(44);
  p.outb = F(45); p.outw = F(46);
  p.tab_tc = F(47); p.tab_turn = F(48);
  return p;
}

extern "C" void kernel_launch(void* const* d_in, const int* in_sizes, int n_in,
                              void* d_out, int out_size, void* d_ws, size_t ws_size,
                              hipStream_t stream) {
  (void)in_sizes; (void)n_in; (void)out_size; (void)ws_size;
  auto F = [&](int i) { return (const float*)d_in[i]; };
  auto I = [&](int i) { return (const int*)d_in[i]; };

  const float* y_hat     = F(0);
  const float* ego_embed = F(1);
  const float* rotl      = F(2);
  const float* orig      = F(3);
  const float* lane_pos  = F(4);
  const float* lane_vec  = F(5);
  const int*   iint      = I(6);
  const int*   iturn     = I(7);
  const int*   itc       = I(8);

  const float* dec_l1b = F(9);   const float* dec_l1w = F(10);
  const float* dec_l2b = F(11);  const float* dec_l2w = F(12);
  const float* dec_lnb = F(13);  const float* dec_lng = F(14);
  EncP enc0 = make_enc(d_in, 15);
  EncP enc1 = make_enc(d_in, 64);
  const float* fcb = F(113); const float* fcw = F(114);
  const float* g_bhh = F(115); const float* g_bih = F(116);
  const float* g_whh = F(117); const float* g_wih = F(118);
  const float* pi_l1b = F(119); const float* pi_l1w = F(120);
  const float* pi_l2b = F(121); const float* pi_l2w = F(122);
  const float* pi_lnb = F(123); const float* pi_lng = F(124);
  const float* pe = F(125);
  const float* sc_l1b = F(126); const float* sc_l1w = F(127);
  const float* sc_l2b = F(128); const float* sc_l2w = F(129);
  const float* sc_lnb = F(130); const float* sc_lng = F(131);

  float* ws    = (float*)d_ws;
  float* ego   = ws;                       // 576*64
  float* emb   = ego   + MN * 64;          // 3*576*64
  float* yinit = emb   + 3 * MN * 64;      // 576*60
  float* delta = yinit + MN * TT * 2;      // 576*60
  float* geom  = delta + MN * TT * 2;      // 576*2*8

  float* out    = (float*)d_out;
  float* trajs  = out;                     // 2*576*30*4 = 138240
  float* pis    = out + 138240;            // 2*576
  float* scores = out + 139392;            // 3*576

  fc_kernel<<<MN, 64, 0, stream>>>(ego_embed, fcw, fcb, ego, emb);
  gru_score_kernel<<<MN, 64, 0, stream>>>(emb, 1, g_wih, g_whh, g_bih, g_bhh,
      sc_l1w, sc_l1b, sc_lng, sc_lnb, sc_l2w, sc_l2b, pe + 0, scores + 0);
  copy_kernel<<<(MN*TT*2 + 255) / 256, 256, 0, stream>>>(y_hat, yinit, MN * TT * 2);

  const float radii[2] = {0.8f, 0.4f};
  for (int r = 0; r < 2; ++r) {
    if (r == 1)
      axpy_kernel<<<(MN*TT*2 + 255) / 256, 256, 0, stream>>>(yinit, delta, MN * TT * 2);
    geom_kernel<<<(MN + 63) / 64, 64, 0, stream>>>(yinit, rotl, orig, radii[r], geom);
    for (int s2 = 0; s2 < 2; ++s2) {
      EncP ep = s2 ? enc1 : enc0;
      enc_kernel<<<MN, TPB, 0, stream>>>(ep, ego, geom, s2, lane_pos, lane_vec,
                                         iint, iturn, itc);
      dec_kernel<<<MN, 64, 0, stream>>>(ego, pe + (r + 1) * 64,
          dec_l1w, dec_l1b, dec_lng, dec_lnb, dec_l2w, dec_l2b,
          trajs + r * (MN * TT * 4), delta, s2);
    }
    head1_kernel<<<MN, 64, 0, stream>>>(ego, pe + (r + 1) * 64,
        pi_l1w, pi_l1b, pi_lng, pi_lnb, pi_l2w, pi_l2b, pis + r * MN);
    copy_kernel<<<(MN*64 + 255) / 256, 256, 0, stream>>>(ego, emb + (r + 1) * MN * 64, MN * 64);
    gru_score_kernel<<<MN, 64, 0, stream>>>(emb, r + 2, g_wih, g_whh, g_bih, g_bhh,
        sc_l1w, sc_l1b, sc_lng, sc_lnb, sc_l2w, sc_l2b,
        pe + (r + 1) * 64, scores + (r + 1) * MN);
  }
}
